// SAttention_584115552312
// MI455X (gfx1250) — matrix operand order repaired and checked by reading, not compiled, not run on hardware
//
#include <hip/hip_runtime.h>
#include <hip/hip_bf16.h>

// ---------------------------------------------------------------------------
// Transformer block (LN -> QKV -> 4-head attention over S -> residual -> LN
// -> FFN -> residual) for x[512,128,256] on MI455X (gfx1250, wave32).
// All matmuls use v_wmma_f32_16x16x32_bf16 (bf16 in, f32 accumulate).
// Global<->LDS staging is fully 128-bit vectorized.
// ---------------------------------------------------------------------------

typedef __attribute__((ext_vector_type(16))) __bf16 v16bf;
typedef __attribute__((ext_vector_type(8)))  float  v8f;

union BF16Frag { v16bf v; unsigned short u[16]; };
union F32x8    { v8f   v; float          f[8];  };

__device__ __forceinline__ unsigned short f2bf(float f) {
    unsigned int u = __float_as_uint(f);
    unsigned int r = u + 0x7fffu + ((u >> 16) & 1u);   // round-to-nearest-even
    return (unsigned short)(r >> 16);
}

__device__ __forceinline__ v8f wmma_bf16(v16bf a, v16bf b, v8f c) {
    // (neg_a, A, neg_b, B, c_mod, C, reuse_a, reuse_b)
    return __builtin_amdgcn_wmma_f32_16x16x32_bf16(false, a, false, b,
                                                   (short)0, c, false, false);
}

// A fragment: 16x32 bf16 tile, row-major in LDS with given stride.
// Contiguous k order (same as B loaders): lane m=L&15, half=L>>4; VGPR v holds K pair
//   k = 2v + 16*half
__device__ __forceinline__ v16bf load_a_frag(const unsigned short* lds,
                                             int stride, int lane) {
    BF16Frag f;
    const int m = lane & 15, hlf = lane >> 4;
#pragma unroll
    for (int v = 0; v < 8; ++v) {
        const int k = 2 * v + 16 * hlf;
        f.u[2 * v]     = lds[m * stride + k];
        f.u[2 * v + 1] = lds[m * stride + k + 1];
    }
    return f.v;
}

// B fragment: 32x16 bf16, memory holds B[k][n] row-major with given stride.
// Contiguous k order (same as A loader): lane n=L&15, half=L>>4; VGPR v holds K pair k = 2v + 16*half
__device__ __forceinline__ v16bf load_b_frag(const unsigned short* lds,
                                             int stride, int lane) {
    BF16Frag f;
    const int n = lane & 15, hlf = lane >> 4;
#pragma unroll
    for (int v = 0; v < 8; ++v) {
        const int k = 2 * v + 16 * hlf;
        f.u[2 * v]     = lds[k * stride + n];
        f.u[2 * v + 1] = lds[(k + 1) * stride + n];
    }
    return f.v;
}

// B fragment where memory holds the TRANSPOSE: B[k][n] = mem[n*stride + k]
// (contiguous per lane -> two ds_load_b128). Used for Q*K^T and weight tiles.
__device__ __forceinline__ v16bf load_bT_frag(const unsigned short* lds,
                                              int stride, int lane) {
    BF16Frag f;
    const int n = lane & 15, hlf = lane >> 4;
#pragma unroll
    for (int v = 0; v < 8; ++v) {
        const int k = 2 * v + 16 * hlf;
        f.u[2 * v]     = lds[n * stride + k];
        f.u[2 * v + 1] = lds[n * stride + k + 1];
    }
    return f.v;
}

// ---------------------------------------------------------------------------
// f32 -> bf16 convert (weights): 8 floats -> 8 bf16 per thread, b128 stores
// ---------------------------------------------------------------------------
__global__ void f32_to_bf16_kernel(const float* __restrict__ in,
                                   unsigned short* __restrict__ out, int n) {
    int i = (blockIdx.x * blockDim.x + threadIdx.x) * 8;
    if (i + 7 < n) {
        const float4 a = *(const float4*)&in[i];
        const float4 b = *(const float4*)&in[i + 4];
        unsigned short r[8];
        r[0] = f2bf(a.x); r[1] = f2bf(a.y); r[2] = f2bf(a.z); r[3] = f2bf(a.w);
        r[4] = f2bf(b.x); r[5] = f2bf(b.y); r[6] = f2bf(b.z); r[7] = f2bf(b.w);
        *(uint4*)&out[i] = *(const uint4*)r;
    }
}

// ---------------------------------------------------------------------------
// LayerNorm over D=256, one wave per row, output bf16. 8 rows / 256-thr block.
// Lane owns 8 contiguous columns -> float4 loads, b128 store.
// ---------------------------------------------------------------------------
__global__ __launch_bounds__(256)
void layernorm_bf16_kernel(const float* __restrict__ X,
                           const float* __restrict__ g,
                           const float* __restrict__ b,
                           unsigned short* __restrict__ Out) {
    const int row  = blockIdx.x * 8 + (threadIdx.x >> 5);
    const int lane = threadIdx.x & 31;
    const int c0   = lane * 8;
    const float* x = X + (size_t)row * 256 + c0;
    float v[8];
    *(float4*)&v[0] = *(const float4*)&x[0];
    *(float4*)&v[4] = *(const float4*)&x[4];
    float s = 0.f;
#pragma unroll
    for (int i = 0; i < 8; ++i) s += v[i];
#pragma unroll
    for (int off = 16; off > 0; off >>= 1) s += __shfl_xor(s, off, 32);
    const float mu = s * (1.0f / 256.0f);
    float q = 0.f;
#pragma unroll
    for (int i = 0; i < 8; ++i) { float d = v[i] - mu; q += d * d; }
#pragma unroll
    for (int off = 16; off > 0; off >>= 1) q += __shfl_xor(q, off, 32);
    const float rstd = rsqrtf(q * (1.0f / 256.0f) + 1e-5f);
    float gg[8], bb[8];
    *(float4*)&gg[0] = *(const float4*)&g[c0];
    *(float4*)&gg[4] = *(const float4*)&g[c0 + 4];
    *(float4*)&bb[0] = *(const float4*)&b[c0];
    *(float4*)&bb[4] = *(const float4*)&b[c0 + 4];
    unsigned short r[8];
#pragma unroll
    for (int i = 0; i < 8; ++i)
        r[i] = f2bf((v[i] - mu) * rstd * gg[i] + bb[i]);
    *(uint4*)&Out[(size_t)row * 256 + c0] = *(const uint4*)r;
}

// ---------------------------------------------------------------------------
// GEMM: Out[M,256] = act( A[M,256] @ W[256,256]^T + bias ) (+ residual)
// A bf16, W bf16 (row n = output feature). 256 threads = 8 waves.
// Block tile: 128(M) x 64(N); each wave owns a 16x64 strip -> 4 f32 accums.
// Staging: b128 loads; weight tile kept in natural [n][k] layout and
// consumed via the transpose-aware fragment loader.
// ---------------------------------------------------------------------------
template <bool RELU, bool BIAS, bool RESID, bool OUT_BF16>
__global__ __launch_bounds__(256)
void gemm256_kernel(const unsigned short* __restrict__ A,
                    const unsigned short* __restrict__ W,
                    const float* __restrict__ bias,
                    const float* __restrict__ resid,
                    void* __restrict__ Out) {
    constexpr int K = 256;
    __shared__ unsigned short sA[128 * 32];   // [m][k]
    __shared__ unsigned short sB[64 * 32];    // [n][k]  (transposed use)

    const int tid  = threadIdx.x;
    const int lane = tid & 31, wave = tid >> 5;
    const int m0   = blockIdx.x * 128;
    const int n0   = blockIdx.y * 64;

    v8f acc[4];
#pragma unroll
    for (int i = 0; i < 4; ++i) acc[i] = (v8f){};

    for (int k0 = 0; k0 < K; k0 += 32) {
        __syncthreads();
        // A tile 128x32: 512 uint4 / 256 thr = 2 each (contiguous along k)
#pragma unroll
        for (int i = 0; i < 2; ++i) {
            const int vi = tid + i * 256;           // 0..511
            const int r = vi >> 2, cv = vi & 3;     // 4 vectors per 32-k row
            *(uint4*)&sA[r * 32 + cv * 8] =
                *(const uint4*)&A[(size_t)(m0 + r) * K + k0 + cv * 8];
        }
        // W tile 64x32 in [n][k]: 256 uint4 / 256 thr = 1 each
        {
            const int nn = tid >> 2, cv = tid & 3;
            *(uint4*)&sB[nn * 32 + cv * 8] =
                *(const uint4*)&W[(size_t)(n0 + nn) * K + k0 + cv * 8];
        }
        __syncthreads();

        const v16bf af = load_a_frag(sA + wave * 16 * 32, 32, lane);
#pragma unroll
        for (int ns = 0; ns < 4; ++ns) {
            const v16bf bf = load_bT_frag(sB + ns * 16 * 32, 32, lane);
            acc[ns] = wmma_bf16(af, bf, acc[ns]);
        }
    }

    // Epilogue. C layout: element r -> row m0+wave*16 + r + 8*(lane>>4),
    //                     col n0 + ns*16 + (lane&15)
    const int hlf = lane >> 4, nloc = lane & 15;
#pragma unroll
    for (int ns = 0; ns < 4; ++ns) {
        F32x8 c; c.v = acc[ns];
#pragma unroll
        for (int r = 0; r < 8; ++r) {
            const int row = m0 + wave * 16 + r + 8 * hlf;
            const int col = n0 + ns * 16 + nloc;
            float val = c.f[r];
            if (BIAS)  val += bias[col];
            if (RELU)  val = fmaxf(val, 0.0f);
            if (RESID) val += resid[(size_t)row * 256 + col];
            if (OUT_BF16)
                ((unsigned short*)Out)[(size_t)row * 256 + col] = f2bf(val);
            else
                ((float*)Out)[(size_t)row * 256 + col] = val;
        }
    }
}

// ---------------------------------------------------------------------------
// Flash attention for one (t, h, s-block): 128 query rows, online softmax
// over all 512 keys, head_dim 64, scale 1/8. Writes xt = x + att to Out.
// 256 threads = 8 waves; each wave owns 16 query rows.
// ---------------------------------------------------------------------------
__global__ __launch_bounds__(256)
void attention_kernel(const unsigned short* __restrict__ Q,
                      const unsigned short* __restrict__ Kb,
                      const unsigned short* __restrict__ Vb,
                      const float* __restrict__ X,
                      float* __restrict__ Out) {
    constexpr int S = 512, T = 128, D = 256, HD = 64;
    const int sb = blockIdx.x;   // query block (0..3), 128 rows each
    const int h  = blockIdx.y;   // head (0..3)
    const int t  = blockIdx.z;   // time (0..127)
    const int tid = threadIdx.x, lane = tid & 31, wave = tid >> 5;
    const int hlf = lane >> 4, nloc = lane & 15;
    const int s0 = sb * 128;

    __shared__ unsigned short sQ[128 * 64];   // 16 KB
    __shared__ unsigned short sK[64 * 64];    //  8 KB
    __shared__ unsigned short sV[64 * 64];    //  8 KB
    __shared__ unsigned short sP[8][16 * 64]; // 16 KB (per-wave P scratch)

    // Stage the 128x64 Q block: 1024 uint4 / 256 thr = 4 each
#pragma unroll
    for (int i = 0; i < 4; ++i) {
        const int vi = tid + i * 256;            // 0..1023
        const int r = vi >> 3, cv = vi & 7;      // 8 vectors per 64-col row
        *(uint4*)&sQ[r * 64 + cv * 8] =
            *(const uint4*)&Q[((size_t)(s0 + r) * T + t) * D + h * HD + cv * 8];
    }
    __syncthreads();

    // Per-wave Q fragments (16 rows x 64 = two k-steps of 32), reused all loop
    const v16bf qf0 = load_a_frag(sQ + wave * 16 * 64, 64, lane);
    const v16bf qf1 = load_a_frag(sQ + wave * 16 * 64 + 32, 64, lane);

    F32x8 o[4];
    float mrow[8], lrow[8];
#pragma unroll
    for (int ns = 0; ns < 4; ++ns)
#pragma unroll
        for (int r = 0; r < 8; ++r) o[ns].f[r] = 0.f;
#pragma unroll
    for (int r = 0; r < 8; ++r) { mrow[r] = -1e30f; lrow[r] = 0.f; }

    for (int j0 = 0; j0 < S; j0 += 64) {
        __syncthreads();
        // Stage K,V tiles 64x64: 512 uint4 each / 256 thr = 2 each
#pragma unroll
        for (int i = 0; i < 2; ++i) {
            const int vi = tid + i * 256;        // 0..511
            const int r = vi >> 3, cv = vi & 7;
            const size_t gidx =
                ((size_t)(j0 + r) * T + t) * D + h * HD + cv * 8;
            *(uint4*)&sK[r * 64 + cv * 8] = *(const uint4*)&Kb[gidx];
            *(uint4*)&sV[r * 64 + cv * 8] = *(const uint4*)&Vb[gidx];
        }
        __syncthreads();

        // Prefetch next j-block K/V into cache hierarchy while we compute
        if (j0 + 64 < S) {
            const int pr = tid >> 2, pc = (tid & 3) * 16;
            const size_t gpre =
                ((size_t)(j0 + 64 + pr) * T + t) * D + h * HD + pc;
            __builtin_prefetch(&Kb[gpre], 0, 3);
            __builtin_prefetch(&Vb[gpre], 0, 3);
        }

        // scores = Q * K^T / 8 : four 16x16 fragments
        F32x8 sc[4];
#pragma unroll
        for (int ns = 0; ns < 4; ++ns) {
            v8f a = (v8f){};
            const v16bf b0 = load_bT_frag(sK + ns * 16 * 64, 64, lane);
            const v16bf b1 = load_bT_frag(sK + ns * 16 * 64 + 32, 64, lane);
            a = wmma_bf16(qf0, b0, a);
            a = wmma_bf16(qf1, b1, a);
            sc[ns].v = a;
        }

        // Online softmax: row max across the 64 columns held by the 16-lane group
        float alpha[8];
#pragma unroll
        for (int r = 0; r < 8; ++r) {
            float mx = mrow[r];
#pragma unroll
            for (int ns = 0; ns < 4; ++ns) {
                sc[ns].f[r] *= 0.125f;               // 1/TEMPERATURE
                mx = fmaxf(mx, sc[ns].f[r]);
            }
#pragma unroll
            for (int off = 1; off < 16; off <<= 1)
                mx = fmaxf(mx, __shfl_xor(mx, off, 32));
            alpha[r] = __expf(mrow[r] - mx);
            mrow[r] = mx;
            lrow[r] *= alpha[r];
        }
#pragma unroll
        for (int ns = 0; ns < 4; ++ns)
#pragma unroll
            for (int r = 0; r < 8; ++r) o[ns].f[r] *= alpha[r];

        // p = exp(s - m); accumulate row sums
#pragma unroll
        for (int r = 0; r < 8; ++r) {
            float psum = 0.f;
#pragma unroll
            for (int ns = 0; ns < 4; ++ns) {
                const float p = __expf(sc[ns].f[r] - mrow[r]);
                sc[ns].f[r] = p;
                psum += p;
            }
#pragma unroll
            for (int off = 1; off < 16; off <<= 1)
                psum += __shfl_xor(psum, off, 32);
            lrow[r] += psum;
        }

        // Re-layout P (C-frag) -> A-frag via per-wave LDS scratch (bf16)
        unsigned short* pw = sP[wave];
#pragma unroll
        for (int ns = 0; ns < 4; ++ns)
#pragma unroll
            for (int r = 0; r < 8; ++r)
                pw[(r + 8 * hlf) * 64 + ns * 16 + nloc] = f2bf(sc[ns].f[r]);
        // wave-local LDS RAW: drain DS counter before fragment reload
        asm volatile("s_wait_dscnt 0" ::: "memory");
        const v16bf pf0 = load_a_frag(pw, 64, lane);
        const v16bf pf1 = load_a_frag(pw + 32, 64, lane);

        // O += P * V  (V tile is [k][n] layout directly)
#pragma unroll
        for (int ns = 0; ns < 4; ++ns) {
            const v16bf v0 = load_b_frag(sV + ns * 16, 64, lane);
            const v16bf v1 = load_b_frag(sV + 32 * 64 + ns * 16, 64, lane);
            o[ns].v = wmma_bf16(pf0, v0, o[ns].v);
            o[ns].v = wmma_bf16(pf1, v1, o[ns].v);
        }
    }

    // Epilogue: xt = x + O / l
#pragma unroll
    for (int ns = 0; ns < 4; ++ns) {
#pragma unroll
        for (int r = 0; r < 8; ++r) {
            const int srow = s0 + wave * 16 + r + 8 * hlf;
            const int col  = h * HD + ns * 16 + nloc;
            const size_t gi = ((size_t)srow * T + t) * D + col;
            Out[gi] = X[gi] + o[ns].f[r] / lrow[r];
        }
    }
}

// ---------------------------------------------------------------------------
// Host launcher
// ---------------------------------------------------------------------------
extern "C" void kernel_launch(void* const* d_in, const int* in_sizes, int n_in,
                              void* d_out, int out_size, void* d_ws, size_t ws_size,
                              hipStream_t stream) {
    (void)in_sizes; (void)n_in; (void)out_size; (void)ws_size;

    const float* x   = (const float*)d_in[0];
    const float* wq  = (const float*)d_in[1];
    const float* wk  = (const float*)d_in[2];
    const float* wv  = (const float*)d_in[3];
    const float* g1  = (const float*)d_in[4];
    const float* be1 = (const float*)d_in[5];
    const float* g2  = (const float*)d_in[6];
    const float* be2 = (const float*)d_in[7];
    const float* w1  = (const float*)d_in[8];
    const float* b1  = (const float*)d_in[9];
    const float* w2  = (const float*)d_in[10];
    const float* b2  = (const float*)d_in[11];
    float* out = (float*)d_out;

    constexpr int    D = 256;
    constexpr size_t NT = 512 * 128;          // tokens
    constexpr size_t WSZ = (size_t)D * D;     // 65536 weight elems

    // Workspace layout (bf16 = ushort). ~129 MB total; xn2/h1 alias xn/q.
    unsigned short* wq_b = (unsigned short*)d_ws;
    unsigned short* wk_b = wq_b + WSZ;
    unsigned short* wv_b = wk_b + WSZ;
    unsigned short* w1_b = wv_b + WSZ;
    unsigned short* w2_b = w1_b + WSZ;
    unsigned short* xn   = w2_b + WSZ;        // [NT, D]
    unsigned short* qb   = xn + NT * D;
    unsigned short* kb   = qb + NT * D;
    unsigned short* vb   = kb + NT * D;
    unsigned short* xn2  = xn;                // reuse after QKV consumed
    unsigned short* h1   = qb;                // reuse after attention consumed

    // 1) weights -> bf16 (8 elems per thread)
    const int cb = (int)((WSZ / 8 + 255) / 256);
    f32_to_bf16_kernel<<<cb, 256, 0, stream>>>(wq, wq_b, (int)WSZ);
    f32_to_bf16_kernel<<<cb, 256, 0, stream>>>(wk, wk_b, (int)WSZ);
    f32_to_bf16_kernel<<<cb, 256, 0, stream>>>(wv, wv_b, (int)WSZ);
    f32_to_bf16_kernel<<<cb, 256, 0, stream>>>(w1, w1_b, (int)WSZ);
    f32_to_bf16_kernel<<<cb, 256, 0, stream>>>(w2, w2_b, (int)WSZ);

    // 2) LN1: x -> xn (bf16)
    layernorm_bf16_kernel<<<NT / 8, 256, 0, stream>>>(x, g1, be1, xn);

    // 3) QKV projections (no bias in reference)
    dim3 ggrid(NT / 128, D / 64);
    gemm256_kernel<false, false, false, true><<<ggrid, 256, 0, stream>>>(xn, wq_b, nullptr, nullptr, qb);
    gemm256_kernel<false, false, false, true><<<ggrid, 256, 0, stream>>>(xn, wk_b, nullptr, nullptr, kb);
    gemm256_kernel<false, false, false, true><<<ggrid, 256, 0, stream>>>(xn, wv_b, nullptr, nullptr, vb);

    // 4) attention + first residual -> d_out holds xt
    attention_kernel<<<dim3(4, 4, 128), 256, 0, stream>>>(qb, kb, vb, x, out);

    // 5) LN2: xt -> xn2 (bf16)
    layernorm_bf16_kernel<<<NT / 8, 256, 0, stream>>>(out, g2, be2, xn2);

    // 6) FFN: h1 = relu(xn2 @ w1^T + b1);  out = xt + h1 @ w2^T + b2
    gemm256_kernel<true,  true, false, true ><<<ggrid, 256, 0, stream>>>(xn2, w1_b, b1, nullptr, h1);
    gemm256_kernel<false, true, true,  false><<<ggrid, 256, 0, stream>>>(h1, w2_b, b2, out, out);
}
